// FeatureFusionLayer_80831284511088
// MI455X (gfx1250) — matrix operand order repaired and checked by reading, not compiled, not run on hardware
//
#include <hip/hip_runtime.h>
#include <hip/hip_bf16.h>
#include <math.h>

// ---------------- problem constants ----------------
constexpr int Bsz = 32, Cch = 3, Tlen = 2048, Fd = 56, Kst = 7, Wwin = 16;
constexpr int T2 = Tlen / Wwin;          // 128
constexpr int NN = 10;
constexpr int Hh = 56;
constexpr int DIN = Cch * Fd * Kst;      // 1176
constexpr int NT  = Bsz * T2;            // 4096 rows
constexpr int G3  = 3 * Hh;              // 168
constexpr int LDK = 1184;                // DIN padded to multiple of 32
constexpr int BROWS = 192;               // B rows padded to multiple of 64 (3 col-blocks)
constexpr float RELIEF_SCALE = 1.0f / (float)(NN * Fd * T2 * Cch); // 1/215040

typedef __attribute__((ext_vector_type(16))) _Float16 v16h;
typedef __attribute__((ext_vector_type(8)))  _Float16 v8h;
typedef __attribute__((ext_vector_type(8)))  float    v8f;
typedef __attribute__((ext_vector_type(4)))  unsigned int u32x4;
typedef __attribute__((ext_vector_type(8)))  int i32x8;
typedef __attribute__((ext_vector_type(4)))  int i32x4;

// ---------------- 1) windowed statistics ----------------
__global__ void k_features(const float* __restrict__ x, float* __restrict__ feat) {
  int tid = blockIdx.x * blockDim.x + threadIdx.x;
  if (tid >= Bsz * Cch * T2 * Fd) return;
  int f  = tid % Fd;
  int t2 = (tid / Fd) % T2;
  int c  = (tid / (Fd * T2)) % Cch;
  int b  =  tid / (Fd * T2 * Cch);
  const float* p = x + (((size_t)(b * Cch + c) * Tlen + (size_t)t2 * Wwin) * Fd + f);

  float v[Wwin];
#pragma unroll
  for (int w = 0; w < Wwin; ++w) v[w] = p[(size_t)w * Fd];

  float amax = v[0], amin = v[0], sum = 0.f;
#pragma unroll
  for (int w = 0; w < Wwin; ++w) { amax = fmaxf(amax, v[w]); amin = fminf(amin, v[w]); sum += v[w]; }
  float mean = sum * (1.0f / Wwin);

  float s2 = 0.f, s3 = 0.f, s4 = 0.f;
#pragma unroll
  for (int w = 0; w < Wwin; ++w) {
    float d = v[w] - mean; float d2 = d * d;
    s2 += d2; s3 += d2 * d; s4 += d2 * d2;
  }
  float stdv = sqrtf(s2 / (float)(Wwin - 1));           // unbiased (torch.std)
  float m2 = s2 * (1.0f / Wwin), m3 = s3 * (1.0f / Wwin), m4 = s4 * (1.0f / Wwin);
  float skew = m3 / (m2 * sqrtf(m2));
  float kurt = m4 / (m2 * m2) - 3.0f;

  // lower-median via stable rank-7 selection (constant indices after unroll)
  float med = 0.f;
#pragma unroll
  for (int i = 0; i < Wwin; ++i) {
    int rank = 0;
#pragma unroll
    for (int j = 0; j < Wwin; ++j)
      rank += (v[j] < v[i]) || (v[j] == v[i] && j < i);
    if (rank == (Wwin - 1) / 2) med = v[i];
  }
  float a[Wwin];
#pragma unroll
  for (int i = 0; i < Wwin; ++i) a[i] = fabsf(v[i] - med);
  float mad = 0.f;
#pragma unroll
  for (int i = 0; i < Wwin; ++i) {
    int rank = 0;
#pragma unroll
    for (int j = 0; j < Wwin; ++j)
      rank += (a[j] < a[i]) || (a[j] == a[i] && j < i);
    if (rank == (Wwin - 1) / 2) mad = a[i];
  }

  float* o = feat + (size_t)tid * Kst;
  o[0] = amax; o[1] = amin; o[2] = mean; o[3] = stdv; o[4] = skew; o[5] = kurt; o[6] = mad;
}

// ---------------- 2) reliefF importance ----------------
__global__ void k_imp_zero(float* __restrict__ imp) {
  if (threadIdx.x < Kst) imp[threadIdx.x] = 0.f;
}

__global__ void __launch_bounds__(64) k_relieff(const float* __restrict__ feat,
                                                float* __restrict__ imp) {
  __shared__ float fs[Fd * Kst];
  __shared__ float rs[Kst];
  const float* fp = feat + (size_t)blockIdx.x * Fd * Kst;
  for (int i = threadIdx.x; i < Fd * Kst; i += blockDim.x) fs[i] = fp[i];
  if (threadIdx.x < Kst) rs[threadIdx.x] = 0.f;
  __syncthreads();

  float sc[Kst] = {0.f, 0.f, 0.f, 0.f, 0.f, 0.f, 0.f};
  const int i = threadIdx.x;
  if (i < Fd) {
    float fi[Kst];
#pragma unroll
    for (int k = 0; k < Kst; ++k) fi[k] = fs[i * Kst + k];
    unsigned long long taken = 0ull;
    for (int p = 0; p < 2 * NN; ++p) {        // first NN: hits (incl self), next NN: misses
      float best = 3.4e38f; int bj = 0;
      for (int j = 0; j < Fd; ++j) {
        if ((taken >> j) & 1ull) continue;
        float d2 = 0.f;
#pragma unroll
        for (int k = 0; k < Kst; ++k) { float df = fi[k] - fs[j * Kst + k]; d2 += df * df; }
        if (d2 < best) { best = d2; bj = j; }  // strict < == stable argsort tie-break
      }
      taken |= 1ull << bj;
      float sgn = (p < NN) ? -1.f : 1.f;
#pragma unroll
      for (int k = 0; k < Kst; ++k) sc[k] += sgn * fabsf(fs[bj * Kst + k] - fi[k]);
    }
  }
  if (i < Fd) {
#pragma unroll
    for (int k = 0; k < Kst; ++k) atomicAdd(&rs[k], sc[k]);
  }
  __syncthreads();
  if (threadIdx.x < Kst) atomicAdd(&imp[threadIdx.x], rs[threadIdx.x] * RELIEF_SCALE);
}

// ---------------- 3) 1x1 conv over flat-view channels, emit f16 seq ----------------
__global__ void k_conv_seq(const float* __restrict__ feat, const float* __restrict__ cw,
                           const float* __restrict__ cb, _Float16* __restrict__ seqh) {
  int tid = blockIdx.x * blockDim.x + threadIdx.x;
  if (tid >= NT * LDK) return;
  int j = tid % LDK; int n = tid / LDK;
  float r = 0.f;
  if (j < DIN) {
    int o = j / (Fd * Kst); int hw = j - o * (Fd * Kst);
    const float* base = feat + (size_t)n * DIN + hw;     // raw flat view, faithful
    r = cw[o * 3 + 0] * base[0] + cw[o * 3 + 1] * base[Fd * Kst]
      + cw[o * 3 + 2] * base[2 * Fd * Kst] + cb[o];
  }
  seqh[tid] = (_Float16)r;   // zero-pads K tail
}

// ---------------- 4) w_ih -> f16 [BROWS][LDK], zero padded ----------------
__global__ void k_cvt_wih(const float* __restrict__ w, _Float16* __restrict__ wh) {
  int tid = blockIdx.x * blockDim.x + threadIdx.x;
  if (tid >= BROWS * LDK) return;
  int k = tid % LDK; int r = tid / LDK;
  float v = (r < G3 && k < DIN) ? w[(size_t)r * DIN + k] : 0.f;
  wh[tid] = (_Float16)v;
}

// -------- 5) WMMA GEMM: each wave computes a 16M x 64N tile (4 wmma / A-load) ------
// A: [M,K] row-major f16 (K padded to ldk); B: [BROWS,K] row-major f16 (zero rows).
// Per-lane fragment addressing follows CDNA5 ISA 7.12.2 16-bit A/B layouts.
__global__ void __launch_bounds__(32) k_wmma_gemm(const _Float16* __restrict__ A,
                                                  const _Float16* __restrict__ Bm,
                                                  const float* __restrict__ bias,
                                                  float* __restrict__ D,
                                                  int ldk, int ncols, int ldc) {
  const int lane = threadIdx.x & 31;
  const int m0 = blockIdx.y * 16;
  const int n0 = blockIdx.x * 64;
  const int half = lane >> 4;
  const int l15 = lane & 15;

  const _Float16* ap = A  + (size_t)(m0 + l15) * ldk + half * 8;   // A: M = l15
  const _Float16* bp = Bm + (size_t)(n0 + l15) * ldk + half * 8;   // B: N = l15 (+16s)

  v8f acc[4] = {v8f{}, v8f{}, v8f{}, v8f{}};
  for (int kb = 0; kb < ldk; kb += 32) {
    if (kb + 256 < ldk) __builtin_prefetch(ap + kb + 256, 0, 0);   // global_prefetch_b8
    v8h alo = *(const v8h*)(ap + kb);        // K = kb+koff..+7     -> V0..V3
    v8h ahi = *(const v8h*)(ap + kb + 16);   // K = kb+16+koff..+7  -> V4..V7
    v16h a = __builtin_shufflevector(alo, ahi, 0, 1, 2, 3, 4, 5, 6, 7,
                                               8, 9, 10, 11, 12, 13, 14, 15);
#pragma unroll
    for (int s = 0; s < 4; ++s) {
      v8h blo = *(const v8h*)(bp + (size_t)s * 16 * ldk + kb);
      v8h bhi = *(const v8h*)(bp + (size_t)s * 16 * ldk + kb + 16);
      v16h b = __builtin_shufflevector(blo, bhi, 0, 1, 2, 3, 4, 5, 6, 7,
                                                 8, 9, 10, 11, 12, 13, 14, 15);
      acc[s] = __builtin_amdgcn_wmma_f32_16x16x32_f16(false, a, false, b,
                                                      (short)0, acc[s], false, false);
    }
  }

  const int mr = m0 + half * 8;
#pragma unroll
  for (int s = 0; s < 4; ++s) {
    const int ncol = n0 + s * 16 + l15;
    if (ncol < ncols) {
      const float bi = bias[ncol];
#pragma unroll
      for (int r = 0; r < 8; ++r)
        D[(size_t)(mr + r) * ldc + ncol] = acc[s][r] + bi;   // C/D: VGPR r -> M=r(+8)
    }
  }
}

// ---------------- 6) GRU recurrence (persistent single workgroup) ----------------
__global__ void __launch_bounds__(256) k_gru(const float* __restrict__ GI,
                                             const float* __restrict__ whh,
                                             const float* __restrict__ bhh,
                                             _Float16* __restrict__ ysh) {
  __shared__ float s_whh[G3 * Hh];   // 37.6 KB
  __shared__ float s_bhh[G3];
  __shared__ float s_h[Bsz * Hh];    // 7 KB
  __shared__ float s_gh[Bsz * G3];   // 21.5 KB
  const int tx = threadIdx.x;

#if __has_builtin(__builtin_amdgcn_tensor_load_to_lds) && __has_builtin(__builtin_amdgcn_s_wait_tensorcnt)
  // Tensor Data Mover: DMA w_hh (9408 f32, one 1-row tile) straight into LDS.
  if (tx < 32) {
    unsigned long long ga = (unsigned long long)(uintptr_t)whh;
    unsigned int lds = (unsigned int)(uintptr_t)(void*)s_whh;
    const unsigned elems = (unsigned)(G3 * Hh);          // 9408
    u32x4 g0;
    g0[0] = 1u;                                          // count=1, user descriptor
    g0[1] = lds;                                         // lds_addr [63:32]
    g0[2] = (unsigned)(ga & 0xffffffffu);                // global_addr lo
    g0[3] = (unsigned)((ga >> 32) & 0x1ffffffu) | (2u << 30);  // addr[56:32] | type=2
    i32x8 g1;
    g1[0] = (int)(2u << 16);                             // workgroup_mask=0, data_size=4B
    g1[1] = (int)((elems & 0xffffu) << 16);              // tensor_dim0[15:0] @ bits63:48
    g1[2] = (int)(((elems >> 16) & 0xffffu) | (1u << 16)); // dim0[31:16] | tensor_dim1=1
    g1[3] = (int)((elems & 0xffffu) << 16);              // dim1 hi=0 | tile_dim0 @127:112
    g1[4] = 1;                                           // tile_dim1=1, tile_dim2=0
    g1[5] = (int)elems;                                  // tensor_dim0_stride lo32
    g1[6] = (int)((elems & 0xffffu) << 16);              // d0stride hi=0 | d1stride lo16
    g1[7] = 0;                                           // d1stride hi
    i32x4 z4 = {0, 0, 0, 0};
    i32x8 z8 = {0, 0, 0, 0, 0, 0, 0, 0};
    __builtin_amdgcn_tensor_load_to_lds(g0, g1, z4, z4, z8, 0);
    __builtin_amdgcn_s_wait_tensorcnt(0);
  }
#else
  for (int i = tx; i < G3 * Hh; i += 256) s_whh[i] = whh[i];
#endif
  for (int i = tx; i < G3; i += 256)       s_bhh[i] = bhh[i];
  for (int i = tx; i < Bsz * Hh; i += 256) s_h[i] = 0.f;
  __syncthreads();

  for (int t = 0; t < T2; ++t) {
    // gh = h @ w_hh^T + b_hh
    for (int i = tx; i < Bsz * G3; i += 256) {
      int row = i % G3; int b = i / G3;
      float s = s_bhh[row];
      const float* hb = &s_h[b * Hh];
      const float* wr = &s_whh[row * Hh];
#pragma unroll 8
      for (int k = 0; k < Hh; ++k) s += hb[k] * wr[k];
      s_gh[i] = s;
    }
    __syncthreads();
    // gates r,z,n + state update; emit f16 ys row (K-padded to 64)
    for (int i = tx; i < Bsz * Hh; i += 256) {
      int j = i % Hh; int b = i / Hh;
      const float* gi = GI + (size_t)(b * T2 + t) * G3;
      const float* gh = &s_gh[b * G3];
      float r = 1.f / (1.f + __expf(-(gi[j] + gh[j])));
      float z = 1.f / (1.f + __expf(-(gi[Hh + j] + gh[Hh + j])));
      float n = tanhf(gi[2 * Hh + j] + r * gh[2 * Hh + j]);
      float hv = (1.f - z) * n + z * s_h[i];
      s_h[i] = hv;
      ysh[(size_t)(b * T2 + t) * 64 + j] = (_Float16)hv;
    }
    for (int i = tx; i < Bsz * 8; i += 256) {     // zero pad cols 56..63
      int b = i / 8; int j = Hh + (i % 8);
      ysh[(size_t)(b * T2 + t) * 64 + j] = (_Float16)0.f;
    }
    __syncthreads();
  }
}

// ---------------- 7) fold imp-weighted K-sum into FC weights ----------------
__global__ void k_foldfc(const float* __restrict__ fcw, const float* __restrict__ fcb,
                         const float* __restrict__ imp, _Float16* __restrict__ wmodh,
                         float* __restrict__ bmod) {
  int tid = blockIdx.x * blockDim.x + threadIdx.x;
  if (tid >= BROWS * 64) return;
  int h = tid % 64; int j = tid / 64;     // j = c*Fd + f -> DIN row = j*7+k
  float v = 0.f;
  if (j < G3 && h < Hh) {
#pragma unroll
    for (int k = 0; k < Kst; ++k) v += imp[k] * fcw[(size_t)(j * Kst + k) * Hh + h];
  }
  wmodh[tid] = (_Float16)v;
  if (h == 0 && j < G3) {
    float s = 0.f;
#pragma unroll
    for (int k = 0; k < Kst; ++k) s += imp[k] * fcb[j * Kst + k];
    bmod[j] = s;
  }
}

// ---------------- launch ----------------
static inline size_t align256(size_t x) { return (x + 255) & ~(size_t)255; }

extern "C" void kernel_launch(void* const* d_in, const int* in_sizes, int n_in,
                              void* d_out, int out_size, void* d_ws, size_t ws_size,
                              hipStream_t stream) {
  (void)in_sizes; (void)n_in; (void)out_size; (void)ws_size;
  const float* x      = (const float*)d_in[0];
  /* y = d_in[1] unused by the reference output */
  const float* conv_w = (const float*)d_in[2];
  const float* conv_b = (const float*)d_in[3];
  const float* w_ih   = (const float*)d_in[4];
  const float* w_hh   = (const float*)d_in[5];
  const float* b_ih   = (const float*)d_in[6];
  const float* b_hh   = (const float*)d_in[7];
  const float* fc_w   = (const float*)d_in[8];
  const float* fc_b   = (const float*)d_in[9];
  float* out = (float*)d_out;                  // [B,T2,C,Fd] = [4096,168]

  // workspace carve (~32.7 MB)
  char* w = (char*)d_ws;
  float*    feat  = (float*)w;    w += align256((size_t)Bsz * Cch * T2 * Fd * Kst * 4);
  _Float16* seqh  = (_Float16*)w; w += align256((size_t)NT * LDK * 2);
  _Float16* wihh  = (_Float16*)w; w += align256((size_t)BROWS * LDK * 2);
  float*    GI    = (float*)w;    w += align256((size_t)NT * G3 * 4);
  _Float16* ysh   = (_Float16*)w; w += align256((size_t)NT * 64 * 2);
  _Float16* wmodh = (_Float16*)w; w += align256((size_t)BROWS * 64 * 2);
  float*    bmod  = (float*)w;    w += align256((size_t)G3 * 4);
  float*    imp   = (float*)w;    w += 256;

  // 1) features
  {
    int n = Bsz * Cch * T2 * Fd;
    k_features<<<(n + 255) / 256, 256, 0, stream>>>(x, feat);
  }
  // 2) reliefF
  k_imp_zero<<<1, 32, 0, stream>>>(imp);
  k_relieff<<<Bsz * Cch * T2, 64, 0, stream>>>(feat, imp);
  // 3) conv -> f16 seq
  {
    int n = NT * LDK;
    k_conv_seq<<<(n + 255) / 256, 256, 0, stream>>>(feat, conv_w, conv_b, seqh);
  }
  // 4) w_ih -> f16 padded
  {
    int n = BROWS * LDK;
    k_cvt_wih<<<(n + 255) / 256, 256, 0, stream>>>(w_ih, wihh);
  }
  // 5) GI = seq @ w_ih^T + b_ih   (WMMA, M=4096 N=168 K=1176)
  {
    dim3 grid(BROWS / 64, NT / 16);
    k_wmma_gemm<<<grid, 32, 0, stream>>>(seqh, wihh, b_ih, GI, LDK, G3, G3);
  }
  // 6) GRU recurrence (sequential over T2, persistent workgroup, w_hh via TDM)
  k_gru<<<1, 256, 0, stream>>>(GI, w_hh, b_hh, ysh);
  // 7) fold imp into fc weights
  {
    int n = BROWS * 64;
    k_foldfc<<<(n + 255) / 256, 256, 0, stream>>>(fc_w, fc_b, imp, wmodh, bmod);
  }
  // 8) out = ys @ W'^T + b'      (WMMA, M=4096 N=168 K=56->64) — writes d_out directly
  {
    dim3 grid(BROWS / 64, NT / 16);
    k_wmma_gemm<<<grid, 32, 0, stream>>>(ysh, wmodh, bmod, out, 64, G3, G3);
  }
}
